// MTGRU_30700426232289
// MI455X (gfx1250) — matrix operand order repaired and checked by reading, not compiled, not run on hardware
//
#include <hip/hip_runtime.h>
#include <hip/hip_bf16.h>

// ---------------------------------------------------------------------------
// MI455X (gfx1250) implementation.
//  * wave32, WMMA 16x16x32 f16->f32 for every dense GEMM.
//  * Recurrences: persistent workgroups, 16-batch-row M-tiles, h in LDS,
//    weights pre-converted to f16 K-contiguous ("B-transposed") layouts.
//  * Per-step input tiles double-buffered in LDS via GLOBAL_LOAD_ASYNC_TO_LDS
//    (ASYNCcnt), exactly one async instruction per wave per step so
//    s_wait_asynccnt 1 implements the double buffer.
//  * Per-batch-element hyper matvecs (different B per row) use coalesced VALU
//    FMAs over f16 Whh_p inside the same persistent decoder workgroup.
// ---------------------------------------------------------------------------

typedef __attribute__((ext_vector_type(16))) _Float16 v16h;
typedef __attribute__((ext_vector_type(8)))  _Float16 v8h;
typedef __attribute__((ext_vector_type(8)))  float    v8f;

union FragU { v16h v; v8h h8[2]; };

#define B_DIM   128
#define T_DIM   1024
#define INP     54
#define HUMAN   64
#define DEC     256
#define ENC     256
#define N_PSI   96540            // 65536 + 256 + 13824 + 16384 + 540
#define N_PSI_PAD 96544          // padded to 16-col tiles (6034 * 16)

// psi per-sample offsets
#define OFF_WHH_P  0
#define OFF_BH_P   65536
#define OFF_WIH_P  65792
#define OFF_C_P    79616
#define OFF_D_P    96000

__device__ __forceinline__ v8f wmma_f16(v16h a, v16h b, v8f c) {
  // emits v_wmma_f32_16x16x32_f16
  return __builtin_amdgcn_wmma_f32_16x16x32_f16(false, a, false, b, (short)0, c,
                                                false, false);
}

// ---- gfx1250 async-copy helpers (ASYNCcnt path) ---------------------------
__device__ __forceinline__ unsigned lds_off_of(const void* p) {
  // LDS aperture: flat addr[31:0] is the workgroup-relative LDS byte offset.
  return (unsigned)(size_t)p;
}
__device__ __forceinline__ void async_ld_b64(unsigned lds_off, const void* g) {
  asm volatile("global_load_async_to_lds_b64 %0, %1, off"
               :: "v"(lds_off), "v"(g) : "memory");
}
__device__ __forceinline__ void async_ld_b128(unsigned lds_off, const void* g) {
  asm volatile("global_load_async_to_lds_b128 %0, %1, off"
               :: "v"(lds_off), "v"(g) : "memory");
}
template <int N>
__device__ __forceinline__ void wait_async() {
  asm volatile("s_wait_asynccnt %0" :: "i"(N) : "memory");
}

// A fragment (16xK tile, row-major f16, rows = M). Lane layout:
// lanes 0-15: M=lane, K = k0+{0..15}; lanes 16-31: M=lane-16,
// K = k0+{16..31}.  Two 16B loads per lane.
__device__ __forceinline__ v16h frag_a(const _Float16* base, int stride, int k0,
                                       int lane) {
  const int m   = lane & 15;
  const int khi = (lane >> 4) << 4;
  const _Float16* p = base + (size_t)m * stride + (size_t)(k0 + khi);
  FragU f;
  f.h8[0] = *(const v8h*)(p);
  f.h8[1] = *(const v8h*)(p + 8);
  return f.v;
}

// B fragment from a K-contiguous ("transposed") weight array W_T[N][K] f16:
// lane column n = n0 + (lane&15); lanes 0-15 hold K=k0..k0+15,
// lanes 16-31 hold K=k0+16..k0+31 (16 contiguous halves per lane).
__device__ __forceinline__ v16h frag_bT(const _Float16* baseT, int stride,
                                        int n0, int k0, int lane) {
  const int n  = n0 + (lane & 15);
  const int ks = k0 + ((lane >> 4) << 4);
  const _Float16* p = baseT + (size_t)n * stride + (size_t)ks;
  FragU f;
  f.h8[0] = *(const v8h*)(p);
  f.h8[1] = *(const v8h*)(p + 8);
  return f.v;
}

// C/D fragment layout: element v of lane l -> M = v + 8*(l>=16), N = l&15.
__device__ __forceinline__ void store_acc_lds_f16(_Float16* g, int stride,
                                                  int n0, v8f acc, int lane) {
  const int n  = n0 + (lane & 15);
  const int mb = (lane >> 4) << 3;
#pragma unroll
  for (int v = 0; v < 8; ++v)
    g[(size_t)(mb + v) * stride + n] = (_Float16)acc[v];
}

__device__ __forceinline__ float sigmoidf_(float x) {
  return 1.f / (1.f + expf(-x));
}

// ---------------------------------------------------------------------------
// Kernel 1: one-time conversions to f16 (K-contiguous layouts, zero pad):
//   Wih16    [768][64]      <- enc_Wih (768,54)
//   Whh16    [768][256]     <- enc_Whh (768,256)
//   gWih16   [512][64]      <- gru_Wih (512,54)
//   gWhhT16  [512][256]     <- gru_Whh (256,512) transposed (gh = h @ Whh)
//   W3_16    [96544][32]    <- W3 (96540,32) (pad rows = 0)
//   inputs16 [B*T][64]      <- inputs (B,T,54) padded, for async staging
// ---------------------------------------------------------------------------
__global__ void convert_weights1(const float* __restrict__ enc_Wih,
                                 const float* __restrict__ enc_Whh,
                                 const float* __restrict__ gru_Wih,
                                 const float* __restrict__ gru_Whh,
                                 const float* __restrict__ W3,
                                 const float* __restrict__ inputs,
                                 _Float16* __restrict__ Wih16,
                                 _Float16* __restrict__ Whh16,
                                 _Float16* __restrict__ gWih16,
                                 _Float16* __restrict__ gWhhT16,
                                 _Float16* __restrict__ W3_16,
                                 _Float16* __restrict__ inputs16) {
  const int N1 = 768 * 64, N2 = 768 * 256, N3 = 512 * 64, N4 = 512 * 256;
  const int N5 = N_PSI_PAD * 32;
  const int N6 = B_DIM * T_DIM * 64;
  const int total = N1 + N2 + N3 + N4 + N5 + N6;
  for (int idx = blockIdx.x * blockDim.x + threadIdx.x; idx < total;
       idx += gridDim.x * blockDim.x) {
    int i = idx;
    if (i < N1) {
      int r = i >> 6, c = i & 63;
      Wih16[i] = (_Float16)((c < INP) ? enc_Wih[r * INP + c] : 0.f);
      continue;
    }
    i -= N1;
    if (i < N2) { Whh16[i] = (_Float16)enc_Whh[i]; continue; }
    i -= N2;
    if (i < N3) {
      int r = i >> 6, c = i & 63;
      gWih16[i] = (_Float16)((c < INP) ? gru_Wih[r * INP + c] : 0.f);
      continue;
    }
    i -= N3;
    if (i < N4) {
      int n = i >> 8, k = i & 255;               // gWhhT16[n][k] = gru_Whh[k][n]
      gWhhT16[i] = (_Float16)gru_Whh[k * 512 + n];
      continue;
    }
    i -= N4;
    if (i < N5) {
      int r = i >> 5, c = i & 31;
      W3_16[i] = (_Float16)((r < N_PSI) ? W3[r * 32 + c] : 0.f);
      continue;
    }
    i -= N5;
    {
      int r = i >> 6, c = i & 63;                // r = b*T + t
      inputs16[i] = (_Float16)((c < INP) ? inputs[(size_t)r * INP + c] : 0.f);
    }
  }
}

// ---------------------------------------------------------------------------
// Kernel 2: encoder GRU.  8 persistent workgroups, each owns 16 batch rows
// for all 1024 steps.  Per step: [16x310] @ [310x768] via WMMA
// (2 k-tiles over padded x, 8 k-tiles over h).  x tiles double-buffered in
// LDS through the async engine (one async b64 per wave per step).
// ---------------------------------------------------------------------------
__global__ __launch_bounds__(256) void encoder_gru_kernel(
    const float* __restrict__ bih, const float* __restrict__ bhh,
    const _Float16* __restrict__ Wih16, const _Float16* __restrict__ Whh16,
    const _Float16* __restrict__ inputs16, float* __restrict__ h_enc) {
  __shared__ __align__(32) _Float16 sh_h16[16 * 256];
  __shared__ __align__(32) float    sh_h32[16 * 256];
  __shared__ __align__(32) _Float16 sh_xbuf[2][16 * 64];  // async staged
  __shared__ __align__(32) _Float16 sh_g[16 * 1024];      // [r|z|inn|hn] x 256

  const int b0 = blockIdx.x * 16;
  const int tid = threadIdx.x, lane = tid & 31, wave = tid >> 5;

  // async staging coordinates for this thread: one 8B chunk of the 16x64 tile
  const int st_row  = tid >> 4;          // 0..15 (batch row in tile)
  const int st_part = (tid & 15) * 4;    // half offset within padded row
  const _Float16* gx_base =
      inputs16 + ((size_t)(b0 + st_row) * T_DIM) * 64 + st_part;

  for (int i = tid; i < 16 * 256; i += 256) {
    sh_h32[i] = 0.f;
    sh_h16[i] = (_Float16)0.f;
  }
  // prologue: stage x(t=0)
  async_ld_b64(lds_off_of(&sh_xbuf[0][st_row * 64 + st_part]), gx_base);
  __syncthreads();

  for (int t = 0; t < T_DIM; ++t) {
    const _Float16* xc = sh_xbuf[t & 1];
    if (t + 1 < T_DIM) {   // stage x(t+1) into the other buffer, then make
      async_ld_b64(        // sure x(t) has landed (oldest of the two).
          lds_off_of(&sh_xbuf[(t + 1) & 1][st_row * 64 + st_part]),
          gx_base + (size_t)(t + 1) * 64);
      wait_async<1>();
    } else {
      wait_async<0>();
    }
    __syncthreads();

    if (wave < 4) {  // r,z gates: gi+gh merge directly (cols 0..511)
      for (int it = 0; it < 8; ++it) {
        const int n0 = (wave * 8 + it) * 16;
        v8f acc = {0.f, 0.f, 0.f, 0.f, 0.f, 0.f, 0.f, 0.f};
#pragma unroll
        for (int kt = 0; kt < 2; ++kt)
          acc = wmma_f16(frag_a(xc, 64, kt * 32, lane),
                         frag_bT(Wih16, 64, n0, kt * 32, lane), acc);
#pragma unroll
        for (int kt = 0; kt < 8; ++kt)
          acc = wmma_f16(frag_a(sh_h16, 256, kt * 32, lane),
                         frag_bT(Whh16, 256, n0, kt * 32, lane), acc);
        store_acc_lds_f16(sh_g, 1024, n0, acc, lane);
      }
    } else {  // n gate: need inn and hn separately (n = tanh(inn + r*hn))
      for (int it = 0; it < 4; ++it) {
        const int n0 = ((wave - 4) * 4 + it) * 16;  // unit column 0..255
        v8f ai = {0.f, 0.f, 0.f, 0.f, 0.f, 0.f, 0.f, 0.f};
        v8f ah = {0.f, 0.f, 0.f, 0.f, 0.f, 0.f, 0.f, 0.f};
#pragma unroll
        for (int kt = 0; kt < 2; ++kt)
          ai = wmma_f16(frag_a(xc, 64, kt * 32, lane),
                        frag_bT(Wih16, 64, 512 + n0, kt * 32, lane), ai);
#pragma unroll
        for (int kt = 0; kt < 8; ++kt)
          ah = wmma_f16(frag_a(sh_h16, 256, kt * 32, lane),
                        frag_bT(Whh16, 256, 512 + n0, kt * 32, lane), ah);
        store_acc_lds_f16(sh_g, 1024, 512 + n0, ai, lane);
        store_acc_lds_f16(sh_g, 1024, 768 + n0, ah, lane);
      }
    }
    __syncthreads();

    const int m = tid >> 4, u0 = (tid & 15) * 16;
#pragma unroll 4
    for (int j = 0; j < 16; ++j) {
      const int u = u0 + j;
      float r = sigmoidf_((float)sh_g[m * 1024 + u] + bih[u] + bhh[u]);
      float z = sigmoidf_((float)sh_g[m * 1024 + 256 + u] + bih[256 + u] +
                          bhh[256 + u]);
      float inn = (float)sh_g[m * 1024 + 512 + u] + bih[512 + u];
      float hn  = (float)sh_g[m * 1024 + 768 + u] + bhh[512 + u];
      float n = tanhf(inn + r * hn);
      float hnew = (1.f - z) * n + z * sh_h32[m * 256 + u];
      sh_h32[m * 256 + u] = hnew;
      sh_h16[m * 256 + u] = (_Float16)hnew;
    }
    __syncthreads();
  }
  for (int i = tid; i < 16 * 256; i += 256)
    h_enc[(size_t)b0 * 256 + i] = sh_h32[i];
}

// ---------------------------------------------------------------------------
// Kernel 3: latent head + psi MLP front (tiny; one block per batch row).
// ---------------------------------------------------------------------------
__global__ __launch_bounds__(64) void latent_kernel(
    const float* __restrict__ h_enc, const float* __restrict__ eps,
    const float* __restrict__ to_mu, const float* __restrict__ to_ls,
    const float* __restrict__ W1, const float* __restrict__ b1,
    const float* __restrict__ W2, const float* __restrict__ b2,
    float* __restrict__ mu_out, float* __restrict__ ls_out,
    _Float16* __restrict__ psi2_16 /*[128][32]*/) {
  const int b = blockIdx.x, tid = threadIdx.x;
  __shared__ float sh_h[256];
  __shared__ float sh_z[16];
  __shared__ float sh_p1[256];
  for (int i = tid; i < 256; i += 64) sh_h[i] = h_enc[(size_t)b * 256 + i];
  __syncthreads();
  if (tid < 16) {
    float m = 0.f, s = 0.f;
    for (int k = 0; k < 256; ++k) {
      m += sh_h[k] * to_mu[tid * 256 + k];
      s += sh_h[k] * to_ls[tid * 256 + k];
    }
    mu_out[b * 16 + tid] = m;
    ls_out[b * 16 + tid] = s;
    sh_z[tid] = m + eps[b * 16 + tid] * expf(s);
  }
  __syncthreads();
  for (int j = tid; j < 256; j += 64) {
    float a = b1[j];
    for (int k = 0; k < 16; ++k) a += sh_z[k] * W1[j * 16 + k];
    sh_p1[j] = tanhf(a);
  }
  __syncthreads();
  if (tid < 32) {
    float a = b2[tid];
    for (int k = 0; k < 256; ++k) a += sh_p1[k] * W2[tid * 256 + k];
    psi2_16[b * 32 + tid] = (_Float16)a;
  }
}

// ---------------------------------------------------------------------------
// Kernel 4: psi = A(128x32) @ W3^T + b3   (WMMA, single K tile of 32).
// 8 waves per block; wave w owns M-tile w; block covers 8 N-tiles.
// ---------------------------------------------------------------------------
__global__ __launch_bounds__(256) void psi_gemm_kernel(
    const _Float16* __restrict__ A16 /*[128][32]*/,
    const _Float16* __restrict__ W3_16 /*[96544][32]*/,
    const float* __restrict__ b3, float* __restrict__ psi /*[128][96540]*/) {
  const int lane = threadIdx.x & 31, wave = threadIdx.x >> 5;
  const int m0 = wave * 16;
  const v16h a = frag_a(A16 + (size_t)m0 * 32, 32, 0, lane);
  for (int i = 0; i < 8; ++i) {
    const int n0 = (blockIdx.x * 8 + i) * 16;
    if (n0 >= N_PSI) break;  // uniform
    if (i + 1 < 8 && n0 + 16 < N_PSI)  // global_prefetch_b8 next B tile
      __builtin_prefetch(W3_16 + (size_t)(n0 + 16 + (lane & 15)) * 32, 0, 0);
    const v16h bm = frag_bT(W3_16, 32, n0, 0, lane);
    v8f acc = {0.f, 0.f, 0.f, 0.f, 0.f, 0.f, 0.f, 0.f};
    acc = wmma_f16(a, bm, acc);
    const int n = n0 + (lane & 15);
    if (n < N_PSI) {
      const float bias = b3[n];
#pragma unroll
      for (int v = 0; v < 8; ++v) {
        const int m = m0 + v + ((lane >> 4) << 3);
        psi[(size_t)m * N_PSI + n] = acc[v] + bias;
      }
    }
  }
}

// ---------------------------------------------------------------------------
// Kernel 5: convert psi slices used as WMMA/VALU-f16 operands:
//   WhhP16 [128][256][256] (straight f16 copy of Whh_p)
//   CpT16  [128][64][256]  (C_p transposed -> K-contiguous B operand)
// ---------------------------------------------------------------------------
__global__ void convert_psi_kernel(const float* __restrict__ psi,
                                   _Float16* __restrict__ WhhP16,
                                   _Float16* __restrict__ CpT16) {
  const int NA = 128 * 65536;
  const int NB = 128 * 16384;
  for (int idx = blockIdx.x * blockDim.x + threadIdx.x; idx < NA + NB;
       idx += gridDim.x * blockDim.x) {
    if (idx < NA) {
      int b = idx >> 16, r = idx & 65535;
      WhhP16[idx] = (_Float16)psi[(size_t)b * N_PSI + OFF_WHH_P + r];
    } else {
      int j = idx - NA;
      int b = j >> 14, r = j & 16383, o = r >> 8, h = r & 255;
      CpT16[j] = (_Float16)psi[(size_t)b * N_PSI + OFF_C_P + h * HUMAN + o];
    }
  }
}

// ---------------------------------------------------------------------------
// Kernel 6: decoder GRU.  8 persistent workgroups x 16 batch rows.
// Shared-weight gates via WMMA; per-sample hyper matvecs (different B per
// batch row -> not WMMA-able) via coalesced VALU FMAs over f16 Whh_p.
// x tiles double-buffered through the async engine as in the encoder.
// ---------------------------------------------------------------------------
__global__ __launch_bounds__(256) void decoder_gru_kernel(
    const float* __restrict__ state, const float* __restrict__ gru_bias,
    const _Float16* __restrict__ gWih16, const _Float16* __restrict__ gWhhT16,
    const float* __restrict__ psi, const _Float16* __restrict__ WhhP16,
    const _Float16* __restrict__ inputs16, _Float16* __restrict__ dec_seq16,
    float* __restrict__ h_last) {
  __shared__ __align__(32) _Float16 sh_h16[16 * 256];
  __shared__ __align__(32) float    sh_h32[16 * 256];
  __shared__ __align__(32) _Float16 sh_xbuf[2][16 * 64];
  __shared__ __align__(32) _Float16 sh_g[16 * 512];  // z|r gate sums

  const int b0 = blockIdx.x * 16;
  const int tid = threadIdx.x, lane = tid & 31, wave = tid >> 5;

  const int st_row  = tid >> 4;
  const int st_part = (tid & 15) * 4;
  const _Float16* gx_base =
      inputs16 + ((size_t)(b0 + st_row) * T_DIM) * 64 + st_part;

  for (int i = tid; i < 16 * 256; i += 256) {
    float v = state[(size_t)b0 * 256 + i];
    sh_h32[i] = v;
    sh_h16[i] = (_Float16)v;
  }
  async_ld_b64(lds_off_of(&sh_xbuf[0][st_row * 64 + st_part]), gx_base);
  __syncthreads();

  const int m = tid >> 4, u0 = (tid & 15) * 16;
  const int b = b0 + m;
  const _Float16* Wp = WhhP16 + (size_t)b * 65536;
  const float* pb = psi + (size_t)b * N_PSI;

  for (int t = 0; t < T_DIM; ++t) {
    const _Float16* xc = sh_xbuf[t & 1];
    if (t + 1 < T_DIM) {
      async_ld_b64(lds_off_of(&sh_xbuf[(t + 1) & 1][st_row * 64 + st_part]),
                   gx_base + (size_t)(t + 1) * 64);
      wait_async<1>();
    } else {
      wait_async<0>();
    }
    __syncthreads();

    // gx + gh (shared weights) -> 32 N-tiles over 8 waves, merged accum.
    for (int it = 0; it < 4; ++it) {
      const int n0 = (wave * 4 + it) * 16;
      v8f acc = {0.f, 0.f, 0.f, 0.f, 0.f, 0.f, 0.f, 0.f};
#pragma unroll
      for (int kt = 0; kt < 2; ++kt)
        acc = wmma_f16(frag_a(xc, 64, kt * 32, lane),
                       frag_bT(gWih16, 64, n0, kt * 32, lane), acc);
#pragma unroll
      for (int kt = 0; kt < 8; ++kt)
        acc = wmma_f16(frag_a(sh_h16, 256, kt * 32, lane),
                       frag_bT(gWhhT16, 256, n0, kt * 32, lane), acc);
      store_acc_lds_f16(sh_g, 512, n0, acc, lane);
    }
    __syncthreads();

    // Per-sample hyper matvec: acc1[u] = bh_p[u] + sum_h h[h]*Whh_p[b][h][u]
    float acc1[16];
#pragma unroll
    for (int j = 0; j < 16; ++j) acc1[j] = pb[OFF_BH_P + u0 + j];
    for (int h = 0; h < 256; ++h) {
      const float hv = sh_h32[m * 256 + h];
      const _Float16* wr = Wp + (size_t)h * 256 + u0;
      FragU w;
      w.h8[0] = *(const v8h*)(wr);
      w.h8[1] = *(const v8h*)(wr + 8);
#pragma unroll
      for (int j = 0; j < 16; ++j) acc1[j] += hv * (float)w.v[j];
    }

    float hnew[16];
#pragma unroll 2
    for (int j = 0; j < 16; ++j) {
      const int u = u0 + j;
      float a0 = 0.f;
      const float* wih = pb + OFF_WIH_P + (size_t)u * INP;
      for (int d = 0; d < INP; ++d) a0 += (float)xc[m * 64 + d] * wih[d];
      float zt = sigmoidf_((float)sh_g[m * 512 + u] + gru_bias[u]);
      float rt = sigmoidf_((float)sh_g[m * 512 + 256 + u] + gru_bias[256 + u]);
      float eta = tanhf(a0 + rt * tanhf(acc1[j]));
      hnew[j] = zt * sh_h32[m * 256 + u] + (1.f - zt) * eta;
    }
    __syncthreads();  // all reads of old h done before overwriting
#pragma unroll
    for (int j = 0; j < 16; ++j) {
      const int u = u0 + j;
      sh_h32[m * 256 + u] = hnew[j];
      sh_h16[m * 256 + u] = (_Float16)hnew[j];
      dec_seq16[((size_t)b * T_DIM + (size_t)t) * 256 + u] = (_Float16)hnew[j];
    }
    __syncthreads();
  }
  for (int i = tid; i < 16 * 256; i += 256)
    h_last[(size_t)b0 * 256 + i] = sh_h32[i];
}

// ---------------------------------------------------------------------------
// Kernel 7: yhats = dec_seq @ C_p + concat(inputs, inputs @ D_p).
// Per block: one (b, 64-row t-tile) -> 4x4 WMMA 16x16 tiles over K=256.
// The contiguous 13.8KB f32 x-tile is staged with async b128 copies.
// ---------------------------------------------------------------------------
__global__ __launch_bounds__(256) void proj_kernel(
    const float* __restrict__ inputs, const _Float16* __restrict__ dec_seq16,
    const _Float16* __restrict__ CpT16, const float* __restrict__ psi,
    float* __restrict__ yhats) {
  __shared__ __align__(16) float sh_x[64 * INP];   // 13824 B, contiguous
  __shared__ float sh_yD[64 * 10];
  const int b = blockIdx.x >> 4;
  const int t0 = (blockIdx.x & 15) * 64;
  const int tid = threadIdx.x, lane = tid & 31, wave = tid >> 5;

  // async-stage the x tile: 864 16B chunks (tile is contiguous in memory)
  const float* xsrc = inputs + ((size_t)b * T_DIM + t0) * INP;
  for (int idx = tid; idx < (64 * INP) / 4; idx += 256)
    async_ld_b128(lds_off_of(&sh_x[idx * 4]), xsrc + idx * 4);
  wait_async<0>();
  __syncthreads();

  const float* Dp = psi + (size_t)b * N_PSI + OFF_D_P;
  for (int i = tid; i < 64 * 10; i += 256) {
    int r = i / 10, e = i % 10;
    float a = 0.f;
    for (int d = 0; d < INP; ++d) a += sh_x[r * INP + d] * Dp[d * 10 + e];
    sh_yD[i] = a;
  }
  __syncthreads();

  const _Float16* A = dec_seq16 + ((size_t)b * T_DIM + t0) * 256;
  const _Float16* Bt = CpT16 + (size_t)b * 64 * 256;
  for (int ti = wave; ti < 16; ti += 8) {  // uniform per wave
    const int mt = ti >> 2, nt = ti & 3;
    v8f acc = {0.f, 0.f, 0.f, 0.f, 0.f, 0.f, 0.f, 0.f};
#pragma unroll
    for (int kt = 0; kt < 8; ++kt)
      acc = wmma_f16(frag_a(A + (size_t)mt * 16 * 256, 256, kt * 32, lane),
                     frag_bT(Bt, 256, nt * 16, kt * 32, lane), acc);
    const int n = nt * 16 + (lane & 15);
#pragma unroll
    for (int v = 0; v < 8; ++v) {
      const int mm = mt * 16 + v + ((lane >> 4) << 3);
      const float add = (n < INP) ? sh_x[mm * INP + n] : sh_yD[mm * 10 + (n - INP)];
      yhats[((size_t)b * T_DIM + t0 + mm) * HUMAN + n] = acc[v] + add;
    }
  }
}

// ---------------------------------------------------------------------------
// Launch
// ---------------------------------------------------------------------------
extern "C" void kernel_launch(void* const* d_in, const int* in_sizes, int n_in,
                              void* d_out, int out_size, void* d_ws,
                              size_t ws_size, hipStream_t stream) {
  (void)in_sizes; (void)n_in; (void)out_size; (void)ws_size;
  const float* inputs   = (const float*)d_in[0];
  const float* state    = (const float*)d_in[1];
  const float* eps      = (const float*)d_in[2];
  const float* enc_Wih  = (const float*)d_in[3];
  const float* enc_Whh  = (const float*)d_in[4];
  const float* enc_bih  = (const float*)d_in[5];
  const float* enc_bhh  = (const float*)d_in[6];
  const float* to_mu    = (const float*)d_in[7];
  const float* to_ls    = (const float*)d_in[8];
  const float* W1       = (const float*)d_in[9];
  const float* b1       = (const float*)d_in[10];
  const float* W2       = (const float*)d_in[11];
  const float* b2       = (const float*)d_in[12];
  const float* W3       = (const float*)d_in[13];
  const float* b3       = (const float*)d_in[14];
  const float* gru_Wih  = (const float*)d_in[15];
  const float* gru_Whh  = (const float*)d_in[16];
  const float* gru_bias = (const float*)d_in[17];

  float* out = (float*)d_out;
  float* yhats  = out;                                  // 128*1024*64
  float* mu_out = out + (size_t)8388608;                // 128*16
  float* ls_out = out + (size_t)8390656;                // 128*16
  float* h_last = out + (size_t)8392704;                // 128*256

  // Workspace layout (byte offsets, 256-aligned)
  char* ws = (char*)d_ws;
  _Float16* Wih16    = (_Float16*)(ws + 0);          //  98304 B
  _Float16* Whh16    = (_Float16*)(ws + 98304);      // 393216 B
  _Float16* gWih16   = (_Float16*)(ws + 491520);     //  65536 B
  _Float16* gWhhT16  = (_Float16*)(ws + 557056);     // 262144 B
  _Float16* W3_16    = (_Float16*)(ws + 819200);     // 6178816 B
  float*    h_enc    = (float*)   (ws + 6998016);    // 131072 B
  _Float16* psi2_16  = (_Float16*)(ws + 7129088);    //   8192 B
  float*    psi      = (float*)   (ws + 7137280);    // 49428480 B
  _Float16* WhhP16   = (_Float16*)(ws + 56565760);   // 16777216 B
  _Float16* CpT16    = (_Float16*)(ws + 73342976);   // 4194304 B
  _Float16* dec_seq  = (_Float16*)(ws + 77537280);   // 67108864 B
  _Float16* inputs16 = (_Float16*)(ws + 144646144);  // 16777216 B (end ~154MB)

  convert_weights1<<<8192, 256, 0, stream>>>(enc_Wih, enc_Whh, gru_Wih, gru_Whh,
                                             W3, inputs, Wih16, Whh16, gWih16,
                                             gWhhT16, W3_16, inputs16);
  encoder_gru_kernel<<<8, 256, 0, stream>>>(enc_bih, enc_bhh, Wih16, Whh16,
                                            inputs16, h_enc);
  latent_kernel<<<128, 64, 0, stream>>>(h_enc, eps, to_mu, to_ls, W1, b1, W2,
                                        b2, mu_out, ls_out, psi2_16);
  psi_gemm_kernel<<<755, 256, 0, stream>>>(psi2_16, W3_16, b3, psi);
  convert_psi_kernel<<<8192, 256, 0, stream>>>(psi, WhhP16, CpT16);
  decoder_gru_kernel<<<8, 256, 0, stream>>>(state, gru_bias, gWih16, gWhhT16,
                                            psi, WhhP16, inputs16, dec_seq,
                                            h_last);
  proj_kernel<<<2048, 256, 0, stream>>>(inputs, dec_seq, CpT16, psi, yhats);
}
